// Decoder_68882685493773
// MI455X (gfx1250) — matrix operand order repaired and checked by reading, not compiled, not run on hardware
//
#include <hip/hip_runtime.h>

// ---------------------------------------------------------------- constants
#define B_   64
#define S_   400
#define T_   100
#define E_   512
#define H_   512
#define E2_  1024   // 2H
#define G_   1536   // 3H
#define XW_  1536   // E + 2H   (GRU input width)
#define PW_  2048   // E + H + 2H (pre-output input width)
#define NEG_ (-1000000000.0f)

typedef __attribute__((ext_vector_type(16))) __bf16 v16bf;
typedef __attribute__((ext_vector_type(8)))  __bf16 v8bf;
typedef __attribute__((ext_vector_type(8)))  float  v8f;

// ---------------------------------------------------------------- fp32 -> bf16
__global__ void k_f32_to_bf16(const float* __restrict__ src,
                              __bf16* __restrict__ dst, int n) {
    int i = blockIdx.x * blockDim.x + threadIdx.x;
    if (i < n) dst[i] = (__bf16)src[i];
}

// ---------------------------------------------------------------- generic WMMA GEMM
// Y[m,n] = act( sum_k X[m,k] * W[n,k] + bias[n] ),  X: MxK bf16, W: NxK bf16,
// Y: f32 with row stride ldY.  Grid: (N/64, M/16), block 128 (4 waves),
// each wave computes one 16x16 tile with a K-loop of v_wmma_f32_16x16x32_bf16.
__global__ void k_wmma_gemm_xwt(const __bf16* __restrict__ X,
                                const __bf16* __restrict__ W,
                                const float*  __restrict__ bias,
                                float* __restrict__ Y,
                                int K, int ldY, int act) {
    int warp  = threadIdx.x >> 5;
    int lane  = threadIdx.x & 31;
    int tileN = blockIdx.x * 4 + warp;
    int tileM = blockIdx.y;
    int half  = lane >> 4;      // which 16-lane half of the wave
    int l16   = lane & 15;

    // A (16x32 bf16, MxK): lane half selects K-subgroup {0..7 / 8..15} (+16 for e>=8)
    const __bf16* xrow = X + (size_t)(tileM * 16 + l16) * K + half * 8;
    // B (32x16 bf16, KxN): same K grouping as A ({0..7 / 8..15} +16 for e>=8)
    const __bf16* wrow = W + (size_t)(tileN * 16 + l16) * K + half * 8;

    v8f acc = {};
    for (int k0 = 0; k0 < K; k0 += 32) {
        v8bf alo = *(const v8bf*)(xrow + k0);        // K = k0 + 8*half + [0..7]
        v8bf ahi = *(const v8bf*)(xrow + k0 + 16);   // K = k0 + 16 + 8*half + [0..7]
        v16bf a = __builtin_shufflevector(alo, ahi,
                    0, 1, 2, 3, 4, 5, 6, 7, 8, 9, 10, 11, 12, 13, 14, 15);
        v8bf blo = *(const v8bf*)(wrow + k0);        // K = k0 + 8*half + [0..7]
        v8bf bhi = *(const v8bf*)(wrow + k0 + 16);   // K = k0 + 16 + 8*half + [0..7]
        v16bf b = __builtin_shufflevector(blo, bhi,
                    0, 1, 2, 3, 4, 5, 6, 7, 8, 9, 10, 11, 12, 13, 14, 15);
        acc = __builtin_amdgcn_wmma_f32_16x16x32_bf16(
                  false, a, false, b, (short)0, acc, false, false);
    }

    int n = tileN * 16 + l16;
    float bv = bias ? bias[n] : 0.0f;
#pragma unroll
    for (int r = 0; r < 8; ++r) {
        int m = tileM * 16 + half * 8 + r;           // C/D layout: VGPR r -> row
        float v = acc[r] + bv;
        if (act == 1) v = tanhf(v);
        Y[(size_t)m * ldY + n] = v;
    }
}

// ---------------------------------------------------------------- attention: scores
// One wave per (b,s): scores[b,s] = sum_h tanh(q[b,h]+proj_key[b,s,h])*w_energy[h]
// Grid: B*S/8 blocks of 256 threads (8 waves) -> 25,600 waves streaming proj_key
// from L2 at full rate.
__global__ void k_scores(const float* __restrict__ q,
                         const float* __restrict__ proj_key,
                         const float* __restrict__ w_energy,
                         const unsigned char* __restrict__ mask,
                         float* __restrict__ scores) {
    int warp = threadIdx.x >> 5, lane = threadIdx.x & 31;
    int idx = blockIdx.x * 8 + warp;      // flat (b*S + s)
    if (idx >= B_ * S_) return;
    int b = idx / S_;
    const float* qb  = q + (size_t)b * H_;
    const float* row = proj_key + (size_t)idx * H_;
    float sum = 0.0f;
    for (int h = lane; h < H_; h += 32)
        sum += tanhf(qb[h] + row[h]) * w_energy[h];
    for (int off = 16; off; off >>= 1) sum += __shfl_xor(sum, off, 32);
    if (lane == 0) scores[idx] = mask[idx] ? sum : NEG_;
}

// ---------------------------------------------------------------- attention: softmax
// One block per b, in-place scores -> alphas over S.
__global__ void k_softmax(float* __restrict__ scores) {
    __shared__ float red[256];
    int b = blockIdx.x, tid = threadIdx.x;
    float* sc = scores + (size_t)b * S_;

    float m = -3.4e38f;
    for (int i = tid; i < S_; i += 256) m = fmaxf(m, sc[i]);
    red[tid] = m; __syncthreads();
    for (int st = 128; st; st >>= 1) { if (tid < st) red[tid] = fmaxf(red[tid], red[tid + st]); __syncthreads(); }
    float mx = red[0]; __syncthreads();

    float e = 0.0f;
    for (int i = tid; i < S_; i += 256) { float ev = __expf(sc[i] - mx); sc[i] = ev; e += ev; }
    red[tid] = e; __syncthreads();
    for (int st = 128; st; st >>= 1) { if (tid < st) red[tid] += red[tid + st]; __syncthreads(); }
    float inv = 1.0f / red[0]; __syncthreads();
    for (int i = tid; i < S_; i += 256) sc[i] *= inv;
}

// ---------------------------------------------------------------- attention: context
// context[b,d] = sum_s alphas[b,s] * enc[b,s,d]
// Grid: (B, 2H/256) blocks of 256 threads; alphas staged in LDS, enc streamed
// coalesced; 256 blocks keep L2 bandwidth saturated.
__global__ void k_context(const float* __restrict__ alphas,
                          const float* __restrict__ enc,
                          float* __restrict__ context) {
    __shared__ float al[S_];
    int b = blockIdx.x, tid = threadIdx.x;
    for (int i = tid; i < S_; i += 256) al[i] = alphas[(size_t)b * S_ + i];
    __syncthreads();

    int d = blockIdx.y * 256 + tid;
    const float* eb = enc + (size_t)b * S_ * E2_ + d;
    float acc = 0.0f;
    for (int s = 0; s < S_; ++s) {
        if ((s & 7) == 0) __builtin_prefetch(eb + (size_t)(s + 8) * E2_, 0, 1);
        acc += al[s] * eb[(size_t)s * E2_];
    }
    context[b * E2_ + d] = acc;
}

// ---------------------------------------------------------------- concat kernels
__global__ void k_xcat(const float* __restrict__ emb,      // trg_embed (B,T,E)
                       const float* __restrict__ context,  // (B, 2H)
                       __bf16* __restrict__ x, int t) {
    int idx = blockIdx.x * blockDim.x + threadIdx.x;
    if (idx >= B_ * XW_) return;
    int b = idx / XW_, j = idx % XW_;
    float v = (j < E_) ? emb[(size_t)b * T_ * E_ + (size_t)t * E_ + j]
                       : context[b * E2_ + (j - E_)];
    x[idx] = (__bf16)v;
}

__global__ void k_precat(const float* __restrict__ emb,
                         const float* __restrict__ h,
                         const float* __restrict__ context,
                         __bf16* __restrict__ p, int t) {
    int idx = blockIdx.x * blockDim.x + threadIdx.x;
    if (idx >= B_ * PW_) return;
    int b = idx / PW_, j = idx % PW_;
    float v;
    if (j < E_)           v = emb[(size_t)b * T_ * E_ + (size_t)t * E_ + j];
    else if (j < E_ + H_) v = h[b * H_ + (j - E_)];
    else                  v = context[b * E2_ + (j - E_ - H_)];
    p[idx] = (__bf16)v;
}

// ---------------------------------------------------------------- GRU gates (in-place h)
__global__ void k_gates(const float* __restrict__ gi, const float* __restrict__ gh,
                        float* __restrict__ h, __bf16* __restrict__ h_bf,
                        float* __restrict__ states, int t) {
    int idx = blockIdx.x * blockDim.x + threadIdx.x;
    if (idx >= B_ * H_) return;
    int b = idx / H_, n = idx % H_;
    float ir = gi[b * G_ + n],           hr = gh[b * G_ + n];
    float iz = gi[b * G_ + H_ + n],      hz = gh[b * G_ + H_ + n];
    float in_ = gi[b * G_ + 2 * H_ + n], hn = gh[b * G_ + 2 * H_ + n];
    float r = 1.0f / (1.0f + __expf(-(ir + hr)));
    float z = 1.0f / (1.0f + __expf(-(iz + hz)));
    float nn = tanhf(in_ + r * hn);
    float hnew = (1.0f - z) * nn + z * h[idx];
    h[idx] = hnew;
    h_bf[idx] = (__bf16)hnew;
    states[(size_t)b * T_ * H_ + (size_t)t * H_ + n] = hnew;
}

__global__ void k_copy(const float* __restrict__ s, float* __restrict__ d, int n) {
    int i = blockIdx.x * blockDim.x + threadIdx.x;
    if (i < n) d[i] = s[i];
}

// ---------------------------------------------------------------- launch
extern "C" void kernel_launch(void* const* d_in, const int* in_sizes, int n_in,
                              void* d_out, int out_size, void* d_ws, size_t ws_size,
                              hipStream_t stream) {
    const float* trg_embed  = (const float*)d_in[0];
    const float* enc_hidden = (const float*)d_in[1];
    const float* enc_final  = (const float*)d_in[2];   // (1,B,2H) == (B,2H)
    const float* W_init     = (const float*)d_in[3];
    const float* b_init     = (const float*)d_in[4];
    const float* W_key      = (const float*)d_in[5];
    const float* W_query    = (const float*)d_in[6];
    const float* W_energy   = (const float*)d_in[7];
    const float* W_ih       = (const float*)d_in[8];
    const float* W_hh       = (const float*)d_in[9];
    const float* b_ih       = (const float*)d_in[10];
    const float* b_hh       = (const float*)d_in[11];
    const float* W_pre      = (const float*)d_in[12];
    const float* b_pre      = (const float*)d_in[13];
    const unsigned char* src_mask = (const unsigned char*)d_in[14];
    (void)in_sizes; (void)n_in; (void)out_size; (void)ws_size;

    // ---- carve workspace (256B aligned chunks)
    char* wsp = (char*)d_ws;
    auto carve = [&](size_t bytes) -> char* {
        char* p = wsp; wsp += (bytes + 255) & ~(size_t)255; return p;
    };
    float*  proj_key = (float*) carve((size_t)B_ * S_ * H_  * 4);
    __bf16* enc_bf   = (__bf16*)carve((size_t)B_ * S_ * E2_ * 2);
    __bf16* efin_bf  = (__bf16*)carve((size_t)B_ * E2_ * 2);
    __bf16* Wq_bf    = (__bf16*)carve((size_t)H_ * H_   * 2);
    __bf16* Wih_bf   = (__bf16*)carve((size_t)G_ * XW_  * 2);
    __bf16* Whh_bf   = (__bf16*)carve((size_t)G_ * H_   * 2);
    __bf16* Wpre_bf  = (__bf16*)carve((size_t)H_ * PW_  * 2);
    __bf16* Wkey_bf  = (__bf16*)carve((size_t)H_ * E2_  * 2);
    __bf16* Winit_bf = (__bf16*)carve((size_t)H_ * E2_  * 2);
    float*  h        = (float*) carve((size_t)B_ * H_  * 4);
    __bf16* h_bf     = (__bf16*)carve((size_t)B_ * H_  * 2);
    float*  q        = (float*) carve((size_t)B_ * H_  * 4);
    float*  scores   = (float*) carve((size_t)B_ * S_  * 4);
    float*  context  = (float*) carve((size_t)B_ * E2_ * 4);
    __bf16* x_bf     = (__bf16*)carve((size_t)B_ * XW_ * 2);
    float*  gi       = (float*) carve((size_t)B_ * G_  * 4);
    float*  gh       = (float*) carve((size_t)B_ * G_  * 4);
    __bf16* pre_bf   = (__bf16*)carve((size_t)B_ * PW_ * 2);

    float* out_states = (float*)d_out;                          // (B,T,H)
    float* out_hlast  = out_states + (size_t)B_ * T_ * H_;      // (1,B,H)
    float* out_pre    = out_hlast + (size_t)B_ * H_;            // (B,T,H)

    auto cvt = [&](const float* s, __bf16* d, size_t n) {
        k_f32_to_bf16<<<dim3((unsigned)((n + 255) / 256)), 256, 0, stream>>>(s, d, (int)n);
    };

    // ---- one-time: bf16 conversions
    cvt(enc_hidden, enc_bf,   (size_t)B_ * S_ * E2_);
    cvt(enc_final,  efin_bf,  (size_t)B_ * E2_);
    cvt(W_query,    Wq_bf,    (size_t)H_ * H_);
    cvt(W_ih,       Wih_bf,   (size_t)G_ * XW_);
    cvt(W_hh,       Whh_bf,   (size_t)G_ * H_);
    cvt(W_pre,      Wpre_bf,  (size_t)H_ * PW_);
    cvt(W_key,      Wkey_bf,  (size_t)H_ * E2_);
    cvt(W_init,     Winit_bf, (size_t)H_ * E2_);

    // ---- hidden0 = tanh(enc_final @ W_init^T + b_init)
    k_wmma_gemm_xwt<<<dim3(H_ / 64, B_ / 16), 128, 0, stream>>>(
        efin_bf, Winit_bf, b_init, h, E2_, H_, /*act=tanh*/1);
    cvt(h, h_bf, (size_t)B_ * H_);

    // ---- proj_key = encoder_hidden @ W_key^T   (25600 x 1024 x 512, L2-resident)
    k_wmma_gemm_xwt<<<dim3(H_ / 64, (B_ * S_) / 16), 128, 0, stream>>>(
        enc_bf, Wkey_bf, nullptr, proj_key, E2_, H_, 0);

    // ---- sequential scan over T steps
    for (int t = 0; t < T_; ++t) {
        // q = h @ W_query^T
        k_wmma_gemm_xwt<<<dim3(H_ / 64, B_ / 16), 128, 0, stream>>>(
            h_bf, Wq_bf, nullptr, q, H_, H_, 0);
        // attention: scores -> softmax -> context (wide parallelism for L2 BW)
        k_scores<<<(B_ * S_) / 8, 256, 0, stream>>>(q, proj_key, W_energy,
                                                    src_mask, scores);
        k_softmax<<<B_, 256, 0, stream>>>(scores);
        k_context<<<dim3(B_, E2_ / 256), 256, 0, stream>>>(scores, enc_hidden, context);
        // x = [prev_embed, context] (bf16)
        k_xcat<<<(B_ * XW_ + 255) / 256, 256, 0, stream>>>(trg_embed, context, x_bf, t);
        // gi = x @ W_ih^T + b_ih ;  gh = h @ W_hh^T + b_hh
        k_wmma_gemm_xwt<<<dim3(G_ / 64, B_ / 16), 128, 0, stream>>>(
            x_bf, Wih_bf, b_ih, gi, XW_, G_, 0);
        k_wmma_gemm_xwt<<<dim3(G_ / 64, B_ / 16), 128, 0, stream>>>(
            h_bf, Whh_bf, b_hh, gh, H_, G_, 0);
        // GRU gates -> h, h_bf, decoder_states[:,t,:]
        k_gates<<<(B_ * H_ + 255) / 256, 256, 0, stream>>>(gi, gh, h, h_bf, out_states, t);
        // pre = [prev_embed, h_new, context] @ W_pre^T + b_pre -> out_pre[:,t,:]
        k_precat<<<(B_ * PW_ + 255) / 256, 256, 0, stream>>>(trg_embed, h, context, pre_bf, t);
        k_wmma_gemm_xwt<<<dim3(H_ / 64, B_ / 16), 128, 0, stream>>>(
            pre_bf, Wpre_bf, b_pre, out_pre + (size_t)t * H_, PW_, T_ * H_, 0);
    }

    // ---- h_last
    k_copy<<<(B_ * H_ + 255) / 256, 256, 0, stream>>>(h, out_hlast, B_ * H_);
}
